// MSDeformableAttention_21148418965573
// MI455X (gfx1250) — matrix operand order repaired and checked by reading, not compiled, not run on hardware
//
#include <hip/hip_runtime.h>
#include <hip/hip_bf16.h>

typedef _Float16 __attribute__((ext_vector_type(8)))  v8h;
typedef _Float16 __attribute__((ext_vector_type(16))) v16h;
typedef float    __attribute__((ext_vector_type(8)))  v8f;

#define BS   16
#define LQ   300
#define DM   256
#define NH   8
#define NC   32
#define NP   16
#define LV   8500
#define NOUT 384                // 256 offset outputs + 128 attn outputs
#define NROW (BS * LQ)          // 4800 query rows

// ---------------------------------------------------------------------------
// Kernel 1: pack query and weights (W_off ++ W_attn) to f16 for WMMA.
// ---------------------------------------------------------------------------
__global__ void msda_pack_f16(const float* __restrict__ q,
                              const float* __restrict__ Woff,
                              const float* __restrict__ Wattn,
                              _Float16* __restrict__ qh,
                              _Float16* __restrict__ wh) {
    int i = blockIdx.x * blockDim.x + threadIdx.x;
    const int NQ = NROW * DM;          // 1,228,800
    const int NW = NOUT * DM;          // 98,304
    if (i < NQ) qh[i] = (_Float16)q[i];
    if (i < NW) {
        float v = (i < 256 * DM) ? Woff[i] : Wattn[i - 256 * DM];
        wh[i] = (_Float16)v;
    }
}

// ---------------------------------------------------------------------------
// Kernel 2: proj = query @ [W_off;W_attn]^T + bias via v_wmma_f32_16x16x32_f16.
// One wave per 16x16 output tile; M=4800 (300 tiles), N=384 (24 tiles), K=256.
// ---------------------------------------------------------------------------
__global__ void msda_proj_wmma(const _Float16* __restrict__ qh,   // [4800][256]
                               const _Float16* __restrict__ wh,   // [384][256]
                               const float* __restrict__ boff,    // [256]
                               const float* __restrict__ battn,   // [128]
                               float* __restrict__ proj) {        // [4800][384]
    const int TILES_N = NOUT / 16;             // 24
    const int NTILES  = (NROW / 16) * TILES_N; // 300*24 = 7200
    int wave = (int)((blockIdx.x * blockDim.x + threadIdx.x) >> 5);
    int lane = (int)(threadIdx.x & 31);
    if (wave >= NTILES) return;                // wave-uniform: EXEC stays full

    int tm = wave / TILES_N;
    int tn = wave % TILES_N;
    int lm = lane & 15;   // row (A) / col (B) within tile
    int hf = lane >> 4;   // half-wave selector per ISA fragment layout

    const _Float16* arow = qh + (size_t)(tm * 16 + lm) * DM;          // A: m = tm*16+lm
    const _Float16* brow = wh + (size_t)(tn * 16 + lm) * DM; // B: n = tn*16+lm

    v8f c = {};
    #pragma unroll
    for (int kc = 0; kc < DM / 32; ++kc) {
        int kb = kc * 32;
        // A 16x32 f16 fragment: VGPR0-3 <-> K = hf*8 + 0..7, VGPR4-7 <-> K+16
        v8h a0 = *(const v8h*)(arow + kb + hf * 8);
        v8h a1 = *(const v8h*)(arow + kb + 16 + hf * 8);
        // B 32x16 f16 fragment: lane n, elements 0-7 <-> K = hf*8 + 0..7, elements 8-15 <-> K+16
        v8h b0 = *(const v8h*)(brow + kb + hf * 8);
        v8h b1 = *(const v8h*)(brow + kb + 16 + hf * 8);
        v16h a, b;
        #pragma unroll
        for (int i = 0; i < 8; ++i) {
            a[i] = a0[i]; a[i + 8] = a1[i];
            b[i] = b0[i]; b[i + 8] = b1[i];
        }
        c = __builtin_amdgcn_wmma_f32_16x16x32_f16(
                /*neg_a=*/false, a, /*neg_b=*/false, b,
                /*c_mod=*/(short)0, c, /*reuse_a=*/false, /*reuse_b=*/false);
    }

    int col = tn * 16 + lm;
    float bias = (col < 256) ? boff[col] : battn[col - 256];
    #pragma unroll
    for (int r = 0; r < 8; ++r) {
        int row = tm * 16 + r + 8 * hf;        // C/D layout: VGPR r, lanes16-31 -> M+8
        proj[(size_t)row * NOUT + col] = c[r] + bias;
    }
}

// ---------------------------------------------------------------------------
// Kernel 3: softmax + bilinear sampling. One wave per (b,q,h); lane = channel.
// ---------------------------------------------------------------------------
__device__ __forceinline__ float msda_tap(const float* __restrict__ vbase,
                                          int lbase, int xi, int yi, int w, int h) {
    bool valid = (xi >= 0) & (xi < w) & (yi >= 0) & (yi < h);
    int xc = min(max(xi, 0), w - 1);
    int yc = min(max(yi, 0), h - 1);
    float g = vbase[(size_t)(lbase + yc * w + xc) * (NH * NC)];
    return valid ? g : 0.0f;
}

__global__ void msda_sample(const float* __restrict__ proj,   // [4800][384]
                            const float* __restrict__ refp,   // [4800][4]
                            const float* __restrict__ value,  // [16][8500][8][32]
                            float* __restrict__ out) {        // [4800][256]
    int wave = (int)((blockIdx.x * blockDim.x + threadIdx.x) >> 5);
    int lane = (int)(threadIdx.x & 31);
    const int TOTAL = NROW * NH;               // 38400
    if (wave >= TOTAL) return;

    int h  = wave % NH;
    int bq = wave / NH;
    int b  = bq / LQ;

    const float* prow = proj + (size_t)bq * NOUT;
    // all 16 point-offsets for this head: one coalesced 32-lane load
    float offv  = prow[h * 32 + lane];
    // attention logits (lanes 0-15 distinct, 16-31 mirror)
    float logit = prow[256 + h * NP + (lane & 15)];

    // softmax over 16 points (width-16 butterfly)
    float mx = logit;
    #pragma unroll
    for (int s = 8; s >= 1; s >>= 1) mx = fmaxf(mx, __shfl_xor(mx, s, 16));
    float e = __expf(logit - mx);
    float sum = e;
    #pragma unroll
    for (int s = 8; s >= 1; s >>= 1) sum += __shfl_xor(sum, s, 16);
    float wght = e / sum;

    const float* ref = refp + (size_t)bq * 4;
    float cx = ref[0], cy = ref[1], rw = ref[2], rh = ref[3];

    const int lw[4]     = {80, 40, 20, 10};
    const int lstart[4] = {0, 6400, 8000, 8400};

    const float* vbase = value + ((size_t)b * LV) * (NH * NC) + h * NC + lane;

    float acc = 0.0f;
    #pragma unroll
    for (int p = 0; p < NP; ++p) {
        float ox = __shfl(offv, 2 * p,     32);
        float oy = __shfl(offv, 2 * p + 1, 32);
        float wp = __shfl(wght, p,         32);
        int l = p >> 2;
        int wd = lw[l];                        // square levels: h == w
        // loc = ref_xy + off * (1/P_l=0.25) * ref_wh * OFFSET_SCALE(0.5)
        float lx = cx + ox * rw * 0.125f;
        float ly = cy + oy * rh * 0.125f;
        float x = lx * (float)wd - 0.5f;
        float y = ly * (float)wd - 0.5f;
        float x0f = floorf(x), y0f = floorf(y);
        int x0 = (int)x0f, y0 = (int)y0f;
        float wx1 = x - x0f, wx0 = 1.0f - wx1;
        float wy1 = y - y0f, wy0 = 1.0f - wy1;
        int lb = lstart[l];
        float g00 = msda_tap(vbase, lb, x0,     y0,     wd, wd);
        float g10 = msda_tap(vbase, lb, x0 + 1, y0,     wd, wd);
        float g01 = msda_tap(vbase, lb, x0,     y0 + 1, wd, wd);
        float g11 = msda_tap(vbase, lb, x0 + 1, y0 + 1, wd, wd);
        acc += wp * (wx0 * wy0 * g00 + wx1 * wy0 * g10 +
                     wx0 * wy1 * g01 + wx1 * wy1 * g11);
    }
    out[(size_t)bq * (NH * NC) + h * NC + lane] = acc;
}

// ---------------------------------------------------------------------------
extern "C" void kernel_launch(void* const* d_in, const int* in_sizes, int n_in,
                              void* d_out, int out_size, void* d_ws, size_t ws_size,
                              hipStream_t stream) {
    const float* query = (const float*)d_in[0];  // [16,300,256]
    const float* refp  = (const float*)d_in[1];  // [16,300,1,4]
    const float* value = (const float*)d_in[2];  // [16,8500,8,32]
    const float* Woff  = (const float*)d_in[3];  // [256,256]
    const float* boff  = (const float*)d_in[4];  // [256]
    const float* Wattn = (const float*)d_in[5];  // [128,256]
    // d_in[6] = b_attn (all zeros in setup, but use it anyway)
    const float* battn = (const float*)d_in[6];  // [128]
    float* out = (float*)d_out;

    // workspace carve-out (all 16B aligned)
    char* ws = (char*)d_ws;
    _Float16* qh   = (_Float16*)(ws);                               // 4800*256*2 = 2,457,600 B
    _Float16* wh   = (_Float16*)(ws + 2457600);                     //  384*256*2 =   196,608 B
    float*    proj = (float*)   (ws + 2457600 + 196608);            // 4800*384*4 = 7,372,800 B

    // 1) pack to f16
    {
        int n = NROW * DM;                      // dominates 384*256
        msda_pack_f16<<<(n + 255) / 256, 256, 0, stream>>>(query, Woff, Wattn, qh, wh);
    }
    // 2) WMMA projection: 7200 tiles, 8 waves/block -> 900 blocks
    {
        int ntiles = (NROW / 16) * (NOUT / 16);
        int blocks = (ntiles * 32 + 255) / 256;
        msda_proj_wmma<<<blocks, 256, 0, stream>>>(qh, wh, boff, battn, proj);
    }
    // 3) softmax + bilinear gather: 38400 waves, 8 waves/block -> 4800 blocks
    {
        int blocks = (NROW * NH * 32 + 255) / 256;
        msda_sample<<<blocks, 256, 0, stream>>>(proj, refp, value, out);
    }
}
